// Recurrence_90623809945678
// MI455X (gfx1250) — matrix operand order repaired and checked by reading, not compiled, not run on hardware
//
#include <hip/hip_runtime.h>
#include <hip/hip_bf16.h>

// ---------------------------------------------------------------------------
// MI455X (gfx1250) implementation.
//   Phase 1: pack weights fp32 -> bf16 WMMA-B-fragment layout (one pass);
//            convert mdp fp32 -> bf16.
//   Phase 2: M = mdp @ emb_W + emb_b          (WMMA bf16 GEMM)
//   Phase 3: bidirectional GRU scan over HR   (WMMA, 16-batch-row tiles, LDS h)
//   Phase 4: T=512 recurrence                 (persistent WGs, WMMA per step)
// All matmuls use v_wmma_f32_16x16x32_bf16 (fp32 accumulate).
// B-matrices pre-packed: each lane's 16 fragment bf16 are contiguous (32B) ->
// two global_load_b128 per fragment. Next-step gather sources are prefetched
// (global_prefetch_b8) to hide gather latency behind the WMMA GEMMs.
// ---------------------------------------------------------------------------

typedef __bf16 bf16_t;
typedef __attribute__((ext_vector_type(16))) __bf16 bf16x16;
typedef __attribute__((ext_vector_type(8)))  __bf16 bf16x8;
typedef __attribute__((ext_vector_type(8)))  float  f32x8;

#define T_STEPS 512
#define NB      256
#define HID     256
#define HRN     64      // mdp rows == n_actions
#define WCN     128
#define VDIM    64
#define H3      768     // 3*H
#define H2      512     // 2*H
#define OUTW    322     // 2 + H + HR

// ---- WMMA fragment helpers (ISA 7.12.2 layouts) ---------------------------

union ABu { bf16x16 v; bf16x8 h[2]; };

// A-frag: 16x32 bf16, row-major source with leading dim lda (16B-aligned rows).
// lane<16: row=lane, K chunks {0..7, 16..23}; lane>=16: K chunks {8..15, 24..31}.
__device__ __forceinline__ bf16x16 load_a_frag(const bf16_t* base, int lda, int lane) {
    const int row = lane & 15;
    const int kb  = (lane >> 4) << 3;
    const bf16_t* p = base + row * lda + kb;
    ABu u;
    u.h[0] = *(const bf16x8*)p;
    u.h[1] = *(const bf16x8*)(p + 16);
    return u.v;
}

// B-frag from pre-packed buffer: tile-major, per-lane 16 contiguous bf16 (32B).
__device__ __forceinline__ bf16x16 load_b_packed(const bf16_t* __restrict__ pk,
                                                 int tile, int lane) {
    return *(const bf16x16*)(pk + (((size_t)tile * 32 + lane) << 4));
}

__device__ __forceinline__ f32x8 wmma_bf16(bf16x16 a, bf16x16 b, f32x8 c) {
    return __builtin_amdgcn_wmma_f32_16x16x32_bf16(false, a, false, b, (short)0, c,
                                                   false, false);
}

__device__ __forceinline__ float sigf(float x) { return 1.f / (1.f + __expf(-x)); }

// tanh via hardware exp: clamp then (e^2x - 1)/(e^2x + 1)
__device__ __forceinline__ float tanhf_fast(float x) {
    x = fminf(fmaxf(x, -15.f), 15.f);
    float e = __expf(2.f * x);
    return (e - 1.f) / (e + 1.f);
}

// ---- Phase 1a: fp32 -> bf16 (mdp) -----------------------------------------

__global__ void cvt_f32_bf16(const float* __restrict__ s, bf16_t* __restrict__ d, int n) {
    int i  = blockIdx.x * blockDim.x + threadIdx.x;
    int st = gridDim.x * blockDim.x;
    for (; i < n; i += st) d[i] = (bf16_t)s[i];
}

// ---- Phase 1b: pack fp32 weight (K x N, row-major) into B-fragment order --
// Packed layout: tile = kt*(N/16)+ct ; lane 0..31 ; 16 contiguous bf16.
// lane<16: col=ct*16+lane, K rows kt*32+{0..7,16..23} ; lane>=16: K rows kt*32+{8..15,24..31}.

__global__ void pack_w_kernel(const float* __restrict__ W, bf16_t* __restrict__ P,
                              int K, int N) {
    const int ctiles = N >> 4;
    const int total  = (K >> 5) * ctiles * 32;
    for (int t = blockIdx.x * blockDim.x + threadIdx.x; t < total;
         t += gridDim.x * blockDim.x) {
        const int lane = t & 31;
        const int tile = t >> 5;
        const int kt = tile / ctiles, ct = tile % ctiles;
        const int col = ct * 16 + (lane & 15);
        const int kb  = kt * 32 + ((lane >> 4) << 3);
        bf16_t* dst = P + (((size_t)tile * 32 + lane) << 4);
#pragma unroll
        for (int e = 0; e < 16; ++e)
            dst[e] = (bf16_t)W[(size_t)(kb + (e & 7) + ((e >> 3) << 4)) * N + col];
    }
}

// ---- Phase 2: M = mdp @ emb_W + emb_b  ->  bf16 (N*HR, H) -----------------

__global__ __launch_bounds__(256, 2)
void embed_kernel(const bf16_t* __restrict__ mdp, const bf16_t* __restrict__ embWp,
                  const float* __restrict__ embB, bf16_t* __restrict__ M) {
    const int lane = threadIdx.x & 31;
    const int wave = threadIdx.x >> 5;
    const int tile = blockIdx.x * 8 + wave;        // 16384 = 16 ct * 1024 rt
    const int rowTiles = (NB * HRN) / 16;          // 1024
    const int ct = tile / rowTiles;
    const int rt = tile % rowTiles;
    const int col = ct * 16 + (lane & 15);

    f32x8 acc;
    const float bv = embB[col];
#pragma unroll
    for (int i = 0; i < 8; ++i) acc[i] = bv;
#pragma unroll
    for (int kt = 0; kt < WCN / 32; ++kt) {
        bf16x16 a = load_a_frag(mdp + (size_t)rt * 16 * WCN + kt * 32, WCN, lane);
        bf16x16 b = load_b_packed(embWp, kt * (HID / 16) + ct, lane);
        acc = wmma_bf16(a, b, acc);
    }
    const int rb = (lane >> 4) << 3;
#pragma unroll
    for (int i = 0; i < 8; ++i) {
        int r = rt * 16 + rb + i;
        M[(size_t)r * HID + col] = (bf16_t)acc[i];
    }
}

// ---- Phase 3: bidirectional GRU scan --------------------------------------
// grid (N/16, 2 dirs), block 128 (4 waves). K(bf16) layout (N, HR, 2H).

__global__ __launch_bounds__(128, 1)
void bigru_kernel(const bf16_t* __restrict__ M,
                  const bf16_t* fwWih, const bf16_t* fwWhh,
                  const float* fwbih, const float* fwbhh,
                  const bf16_t* bwWih, const bf16_t* bwWhh,
                  const float* bwbih, const float* bwbhh,
                  bf16_t* __restrict__ K) {
    extern __shared__ char smem[];
    bf16_t* xs = (bf16_t*)smem;                 // 16x256 bf16
    bf16_t* hb = xs + 16 * HID;                 // 16x256 bf16
    float*  hf = (float*)(hb + 16 * HID);       // 16x256 f32
    float*  gi = hf + 16 * HID;                 // 16x768 f32
    float*  gh = gi + 16 * H3;                  // 16x768 f32

    const int dir  = blockIdx.y;
    const int n0   = blockIdx.x * 16;
    const int tid  = threadIdx.x;
    const int lane = tid & 31;
    const int wave = tid >> 5;

    const bf16_t* Wih = dir ? bwWih : fwWih;    // packed, ctiles = 48
    const bf16_t* Whh = dir ? bwWhh : fwWhh;
    const float*  bih = dir ? bwbih : fwbih;
    const float*  bhh = dir ? bwbhh : fwbhh;

    for (int i = tid; i < 16 * HID; i += 128) { hf[i] = 0.f; hb[i] = (bf16_t)0.f; }
    __syncthreads();

    for (int s = 0; s < HRN; ++s) {
        const int srow = dir ? (HRN - 1 - s) : s;
        // x tile: 16 rows x 256 bf16, 16B-chunk copies
        for (int i = tid; i < 16 * (HID / 8); i += 128) {
            int r = i >> 5, c8 = (i & 31) << 3;
            *(bf16x8*)(xs + r * HID + c8) =
                *(const bf16x8*)(M + ((size_t)(n0 + r) * HRN + srow) * HID + c8);
        }
        __syncthreads();

        // prefetch next seq row (fire-and-forget, overlaps with WMMA below)
        if (s + 1 < HRN && tid < 64) {
            const int nrow = dir ? (HRN - 2 - s) : (s + 1);
            const int r = tid >> 2, seg = tid & 3;
            __builtin_prefetch(M + ((size_t)(n0 + r) * HRN + nrow) * HID + seg * 64,
                               0, 1);
        }

        // gi = x@Wih + bih, gh = h@Whh + bhh : 48 coltiles each, 12 per wave
        for (int cti = 0; cti < 12; ++cti) {
            const int ct  = wave * 12 + cti;
            const int col = ct * 16 + (lane & 15);
            f32x8 ai, ah;
            const float b1 = bih[col], b2 = bhh[col];
#pragma unroll
            for (int i = 0; i < 8; ++i) { ai[i] = b1; ah[i] = b2; }
#pragma unroll
            for (int kt = 0; kt < HID / 32; ++kt) {
                bf16x16 ax = load_a_frag(xs + kt * 32, HID, lane);
                bf16x16 bx = load_b_packed(Wih, kt * 48 + ct, lane);
                ai = wmma_bf16(ax, bx, ai);
                bf16x16 ahh  = load_a_frag(hb + kt * 32, HID, lane);
                bf16x16 bhh2 = load_b_packed(Whh, kt * 48 + ct, lane);
                ah = wmma_bf16(ahh, bhh2, ah);
            }
            const int rb = (lane >> 4) << 3;
#pragma unroll
            for (int i = 0; i < 8; ++i) {
                gi[(rb + i) * H3 + col] = ai[i];
                gh[(rb + i) * H3 + col] = ah[i];
            }
        }
        __syncthreads();

        for (int i = tid; i < 16 * HID; i += 128) {
            int r = i >> 8, c = i & 255;
            float irv = gi[r * H3 + c],            hrv = gh[r * H3 + c];
            float izv = gi[r * H3 + HID + c],      hzv = gh[r * H3 + HID + c];
            float inv = gi[r * H3 + 2 * HID + c],  hnv = gh[r * H3 + 2 * HID + c];
            float rg = sigf(irv + hrv);
            float zg = sigf(izv + hzv);
            float ng = tanhf_fast(inv + rg * hnv);
            float hn = (1.f - zg) * ng + zg * hf[i];
            hf[i] = hn;
            hb[i] = (bf16_t)hn;
            K[((size_t)(n0 + r) * HRN + srow) * H2 + (size_t)dir * HID + c] = (bf16_t)hn;
        }
        __syncthreads();
    }
}

// ---- Phase 4: main T-step recurrence --------------------------------------
// grid N/16 = 16 persistent WGs, block 256 (8 waves).

__global__ __launch_bounds__(256, 1)
void recurrence_kernel(const float* __restrict__ values, const int* __restrict__ actions,
                       const int* __restrict__ a0, const float* __restrict__ h0,
                       const bf16_t* __restrict__ M, const bf16_t* __restrict__ K,
                       const bf16_t* f0Wp, const float* f0b,
                       const bf16_t* f1Wp, const float* f1b,
                       const bf16_t* cWihp, const bf16_t* cWhhp,
                       const float* cbih, const float* cbhh,
                       const float* criticW, const float* criticb,
                       const bf16_t* qgWp, const float* qgb,
                       float* __restrict__ out) {
    extern __shared__ char smem[];
    bf16_t* xin = (bf16_t*)smem;                  // 16x320 bf16
    bf16_t* mid = xin + 16 * 320;                 // 16x256 bf16
    bf16_t* hb  = mid + 16 * HID;                 // 16x256 bf16
    float*  hf  = (float*)(hb + 16 * HID);        // 16x256 f32
    float*  S1  = hf + 16 * HID;                  // 16x768 f32 (gi / query k)
    float*  S2  = S1 + 16 * H3;                   // 16x768 f32 (gh)
    float*  lg  = S2 + 16 * H3;                   // 16x64  f32 logits

    const int n0   = blockIdx.x * 16;
    const int tid  = threadIdx.x;
    const int lane = tid & 31;
    const int wave = tid >> 5;

    for (int i = tid; i < 16 * HID; i += 256) {
        int r = i >> 8, c = i & 255;
        float v = h0[(size_t)(n0 + r) * HID + c];
        hf[i] = v;
        hb[i] = (bf16_t)v;
    }
    __syncthreads();

    for (int t = 0; t < T_STEPS; ++t) {
        // x = [values_t | M[n, a_prev]]
        for (int i = tid; i < 16 * VDIM; i += 256) {
            int r = i >> 6, c = i & 63;
            xin[r * 320 + c] = (bf16_t)values[((size_t)t * NB + n0 + r) * VDIM + c];
        }
        for (int i = tid; i < 16 * (HID / 8); i += 256) {
            int r = i >> 5, c8 = (i & 31) << 3;
            int n = n0 + r;
            int ap = (t == 0) ? a0[n] : actions[(size_t)(t - 1) * NB + n];
            *(bf16x8*)(xin + r * 320 + 64 + c8) =
                *(const bf16x8*)(M + ((size_t)n * HRN + ap) * HID + c8);
        }
        __syncthreads();

        // prefetch next step's gather sources (overlaps with the GEMMs below):
        //   M rows selected by actions[t] (512B each) and values[t+1] (256B each)
        if (t + 1 < T_STEPS) {
            if (tid < 64) {                       // 16 rows x 4 cachelines of M
                const int r = tid >> 2, seg = tid & 3;
                const int apn = actions[(size_t)t * NB + n0 + r];
                __builtin_prefetch(M + ((size_t)(n0 + r) * HRN + apn) * HID + seg * 64,
                                   0, 1);
            } else if (tid < 96) {                // 16 rows x 2 cachelines of values
                const int r = (tid - 64) >> 1, seg = tid & 1;
                __builtin_prefetch(values + ((size_t)(t + 1) * NB + n0 + r) * VDIM
                                        + seg * 32, 0, 1);
            }
        }

        // f0: relu((16x320)@(320x256)+b) -> mid ; 16 coltiles, 2 per wave
        for (int cti = 0; cti < 2; ++cti) {
            const int ct  = wave * 2 + cti;
            const int col = ct * 16 + (lane & 15);
            f32x8 acc; const float bv = f0b[col];
#pragma unroll
            for (int i = 0; i < 8; ++i) acc[i] = bv;
#pragma unroll
            for (int kt = 0; kt < 320 / 32; ++kt) {
                bf16x16 a = load_a_frag(xin + kt * 32, 320, lane);
                bf16x16 b = load_b_packed(f0Wp, kt * 16 + ct, lane);
                acc = wmma_bf16(a, b, acc);
            }
            const int rb = (lane >> 4) << 3;
#pragma unroll
            for (int i = 0; i < 8; ++i)
                mid[(rb + i) * HID + col] = (bf16_t)fmaxf(acc[i], 0.f);
        }
        __syncthreads();

        // f1: relu((16x256)@(256x256)+b) -> xin (ld 320, cols 0..255)
        for (int cti = 0; cti < 2; ++cti) {
            const int ct  = wave * 2 + cti;
            const int col = ct * 16 + (lane & 15);
            f32x8 acc; const float bv = f1b[col];
#pragma unroll
            for (int i = 0; i < 8; ++i) acc[i] = bv;
#pragma unroll
            for (int kt = 0; kt < HID / 32; ++kt) {
                bf16x16 a = load_a_frag(mid + kt * 32, HID, lane);
                bf16x16 b = load_b_packed(f1Wp, kt * 16 + ct, lane);
                acc = wmma_bf16(a, b, acc);
            }
            const int rb = (lane >> 4) << 3;
#pragma unroll
            for (int i = 0; i < 8; ++i)
                xin[(rb + i) * 320 + col] = (bf16_t)fmaxf(acc[i], 0.f);
        }
        __syncthreads();

        // GRU cell: waves 0-3 -> gi = x1@cWih+bih (S1); waves 4-7 -> gh = h@cWhh+bhh (S2)
        {
            const bf16_t* A   = (wave < 4) ? xin : hb;
            const int     lda = (wave < 4) ? 320 : HID;
            const bf16_t* W   = (wave < 4) ? cWihp : cWhhp;
            const float*  bb  = (wave < 4) ? cbih : cbhh;
            float*        Sd  = (wave < 4) ? S1 : S2;
            const int w4 = wave & 3;
            for (int cti = 0; cti < 12; ++cti) {
                const int ct  = w4 * 12 + cti;
                const int col = ct * 16 + (lane & 15);
                f32x8 acc; const float bv = bb[col];
#pragma unroll
                for (int i = 0; i < 8; ++i) acc[i] = bv;
#pragma unroll
                for (int kt = 0; kt < HID / 32; ++kt) {
                    bf16x16 a = load_a_frag(A + kt * 32, lda, lane);
                    bf16x16 b = load_b_packed(W, kt * 48 + ct, lane);
                    acc = wmma_bf16(a, b, acc);
                }
                const int rb = (lane >> 4) << 3;
#pragma unroll
                for (int i = 0; i < 8; ++i) Sd[(rb + i) * H3 + col] = acc[i];
            }
        }
        __syncthreads();

        for (int i = tid; i < 16 * HID; i += 256) {
            int r = i >> 8, c = i & 255;
            float irv = S1[r * H3 + c],           hrv = S2[r * H3 + c];
            float izv = S1[r * H3 + HID + c],     hzv = S2[r * H3 + HID + c];
            float inv = S1[r * H3 + 2 * HID + c], hnv = S2[r * H3 + 2 * HID + c];
            float rg = sigf(irv + hrv);
            float zg = sigf(izv + hzv);
            float ng = tanhf_fast(inv + rg * hnv);
            float hn = (1.f - zg) * ng + zg * hf[i];
            hf[i] = hn;
            hb[i] = (bf16_t)hn;
        }
        __syncthreads();

        // query k = h @ qgW + qgb -> S1 (16x512, ld H2) ; 32 coltiles, 4 per wave
        for (int cti = 0; cti < 4; ++cti) {
            const int ct  = wave * 4 + cti;
            const int col = ct * 16 + (lane & 15);
            f32x8 acc; const float bv = qgb[col];
#pragma unroll
            for (int i = 0; i < 8; ++i) acc[i] = bv;
#pragma unroll
            for (int kt = 0; kt < HID / 32; ++kt) {
                bf16x16 a = load_a_frag(hb + kt * 32, HID, lane);
                bf16x16 b = load_b_packed(qgWp, kt * 32 + ct, lane);
                acc = wmma_bf16(a, b, acc);
            }
            const int rb = (lane >> 4) << 3;
#pragma unroll
            for (int i = 0; i < 8; ++i) S1[(rb + i) * H2 + col] = acc[i];
        }
        __syncthreads();

        // logits[n, s] = K[n, s, :] . k[n, :]  (1024 dots of length 512)
        for (int task = tid; task < 16 * HRN; task += 256) {
            const int r = task >> 6, s = task & 63;
            const bf16_t* Kr = K + ((size_t)(n0 + r) * HRN + s) * H2;
            const float*  kq = S1 + r * H2;
            float acc = 0.f;
#pragma unroll 4
            for (int c = 0; c < H2; c += 8) {
                bf16x8 kv = *(const bf16x8*)(Kr + c);
                float4 q0 = *(const float4*)(kq + c);
                float4 q1 = *(const float4*)(kq + c + 4);
                acc += (float)kv[0] * q0.x + (float)kv[1] * q0.y
                     + (float)kv[2] * q0.z + (float)kv[3] * q0.w
                     + (float)kv[4] * q1.x + (float)kv[5] * q1.y
                     + (float)kv[6] * q1.z + (float)kv[7] * q1.w;
            }
            lg[r * HRN + s] = acc;
        }
        __syncthreads();

        // outputs: hx[t, n] = [a, v, h(256), probs(64)]
        const size_t obase = ((size_t)t * NB + n0) * OUTW;
        if (tid < 16) {
            const int r = tid;
            float mx = -1e30f;
            for (int s = 0; s < HRN; ++s) mx = fmaxf(mx, lg[r * HRN + s]);
            float sum = 0.f;
            for (int s = 0; s < HRN; ++s) {
                float e = __expf(lg[r * HRN + s] - mx);
                lg[r * HRN + s] = e;
                sum += e;
            }
            const float inv = 1.f / sum;
            float* po = out + obase + (size_t)r * OUTW + 258;
            for (int s = 0; s < HRN; ++s) po[s] = lg[r * HRN + s] * inv;
        } else if (tid < 32) {
            const int r = tid - 16;
            float acc = criticb[0];
            for (int c = 0; c < HID; ++c) acc += hf[r * HID + c] * criticW[c];
            out[obase + (size_t)r * OUTW + 1] = acc;
            out[obase + (size_t)r * OUTW + 0] =
                (float)actions[(size_t)t * NB + n0 + r];
        }
        for (int i = tid; i < 16 * HID; i += 256) {
            int r = i >> 8, c = i & 255;
            out[obase + (size_t)r * OUTW + 2 + c] = hf[i];
        }
        __syncthreads();
    }
}

// ---------------------------------------------------------------------------

extern "C" void kernel_launch(void* const* d_in, const int* in_sizes, int n_in,
                              void* d_out, int out_size, void* d_ws, size_t ws_size,
                              hipStream_t stream) {
    const float* values  = (const float*)d_in[0];
    const float* mdp     = (const float*)d_in[1];
    const int*   actions = (const int*)d_in[2];
    const int*   a0      = (const int*)d_in[3];
    const float* h0      = (const float*)d_in[4];
    const float* emb_W   = (const float*)d_in[5];
    const float* emb_b   = (const float*)d_in[6];
    const float* gfw_Wih = (const float*)d_in[7];
    const float* gfw_Whh = (const float*)d_in[8];
    const float* gfw_bih = (const float*)d_in[9];
    const float* gfw_bhh = (const float*)d_in[10];
    const float* gbw_Wih = (const float*)d_in[11];
    const float* gbw_Whh = (const float*)d_in[12];
    const float* gbw_bih = (const float*)d_in[13];
    const float* gbw_bhh = (const float*)d_in[14];
    const float* f0_W    = (const float*)d_in[15];
    const float* f0_b    = (const float*)d_in[16];
    const float* f1_W    = (const float*)d_in[17];
    const float* f1_b    = (const float*)d_in[18];
    const float* c_Wih   = (const float*)d_in[19];
    const float* c_Whh   = (const float*)d_in[20];
    const float* c_bih   = (const float*)d_in[21];
    const float* c_bhh   = (const float*)d_in[22];
    const float* critW   = (const float*)d_in[23];
    const float* critb   = (const float*)d_in[24];
    const float* qg_W    = (const float*)d_in[25];
    const float* qg_b    = (const float*)d_in[26];
    float* out = (float*)d_out;

    // workspace carve (256B aligned)
    char*  ws  = (char*)d_ws;
    size_t off = 0;
    auto carve = [&](size_t elems) -> bf16_t* {
        bf16_t* p = (bf16_t*)(ws + off);
        off = (off + elems * sizeof(bf16_t) + 255) & ~(size_t)255;
        return p;
    };
    bf16_t* mdp_b   = carve((size_t)NB * HRN * WCN);
    bf16_t* embW_p  = carve((size_t)WCN * HID);
    bf16_t* fwWih_p = carve((size_t)HID * H3);
    bf16_t* fwWhh_p = carve((size_t)HID * H3);
    bf16_t* bwWih_p = carve((size_t)HID * H3);
    bf16_t* bwWhh_p = carve((size_t)HID * H3);
    bf16_t* f0W_p   = carve((size_t)(VDIM + HID) * HID);
    bf16_t* f1W_p   = carve((size_t)HID * HID);
    bf16_t* cWih_p  = carve((size_t)HID * H3);
    bf16_t* cWhh_p  = carve((size_t)HID * H3);
    bf16_t* qgW_p   = carve((size_t)HID * H2);
    bf16_t* M_b     = carve((size_t)NB * HRN * HID);
    bf16_t* K_b     = carve((size_t)NB * HRN * H2);
    (void)ws_size; (void)n_in; (void)in_sizes; (void)out_size;

    {
        const int n = NB * HRN * WCN;
        cvt_f32_bf16<<<1024, 256, 0, stream>>>(mdp, mdp_b, n);
    }
    auto pack = [&](const float* s, bf16_t* d, int K, int N) {
        const int total = (K >> 5) * (N >> 4) * 32;
        int blocks = (total + 255) / 256;
        if (blocks > 1024) blocks = 1024;
        pack_w_kernel<<<blocks, 256, 0, stream>>>(s, d, K, N);
    };
    pack(emb_W,   embW_p,  WCN,        HID);
    pack(gfw_Wih, fwWih_p, HID,        H3);
    pack(gfw_Whh, fwWhh_p, HID,        H3);
    pack(gbw_Wih, bwWih_p, HID,        H3);
    pack(gbw_Whh, bwWhh_p, HID,        H3);
    pack(f0_W,    f0W_p,   VDIM + HID, HID);
    pack(f1_W,    f1W_p,   HID,        HID);
    pack(c_Wih,   cWih_p,  HID,        H3);
    pack(c_Whh,   cWhh_p,  HID,        H3);
    pack(qg_W,    qgW_p,   HID,        H2);

    // Phase 2: 16384 16x16 tiles, 8 waves/WG
    embed_kernel<<<2048, 256, 0, stream>>>(mdp_b, embW_p, emb_b, M_b);

    // Phase 3: 16 batch tiles x 2 directions, 128 threads, 128 KB dynamic LDS
    {
        const size_t lds = (size_t)16 * HID * sizeof(bf16_t) * 2   // xs + hb
                         + (size_t)16 * HID * sizeof(float)        // hf
                         + (size_t)16 * H3  * sizeof(float) * 2;   // gi + gh
        dim3 grid(NB / 16, 2);
        bigru_kernel<<<grid, 128, lds, stream>>>(
            M_b, fwWih_p, fwWhh_p, gfw_bih, gfw_bhh,
            bwWih_p, bwWhh_p, gbw_bih, gbw_bhh, K_b);
    }

    // Phase 4: 16 persistent WGs, 256 threads, ~142 KB dynamic LDS
    {
        const size_t lds = (size_t)16 * 320 * sizeof(bf16_t)       // xin
                         + (size_t)16 * HID * sizeof(bf16_t) * 2   // mid + hb
                         + (size_t)16 * HID * sizeof(float)        // hf
                         + (size_t)16 * H3  * sizeof(float) * 2    // S1 + S2
                         + (size_t)16 * HRN * sizeof(float);       // logits
        recurrence_kernel<<<NB / 16, 256, lds, stream>>>(
            values, actions, a0, h0, M_b, K_b,
            f0W_p, f0_b, f1W_p, f1_b,
            cWih_p, cWhh_p, c_bih, c_bhh,
            critW, critb, qgW_p, qg_b, out);
    }
}
